// SSPDLoss_47184510714643
// MI455X (gfx1250) — matrix operand order repaired and checked by reading, not compiled, not run on hardware
//
#include <hip/hip_runtime.h>
#include <math.h>

// ---------------------------------------------------------------------------
// SSPD loss for MI455X (gfx1250, wave32).
// Dominant work: 128-batch 256x256x768 Gram matrices -> v_wmma_f32_16x16x32_bf16
// with a 2-term bf16 split of the fp32 inputs (hi*hi + hi*lo + lo*hi).
// All LDS traffic is vectorized: ds_store_b64 staging, ds_load_b128 gathers.
// Everything else is L2-resident reductions (192MB L2 >> 64MB of ssm data).
// Workspace requirement: ~68 MB of d_ws.
// ---------------------------------------------------------------------------

typedef __attribute__((ext_vector_type(16))) __bf16 v16bf;
typedef __attribute__((ext_vector_type(8)))  float  v8f;

union BfVec { v16bf v; unsigned u[8]; uint4 q[2]; };

static __device__ inline unsigned pack2(__bf16 a, __bf16 b) {
    unsigned short x = __builtin_bit_cast(unsigned short, a);
    unsigned short y = __builtin_bit_cast(unsigned short, b);
    return (unsigned)x | ((unsigned)y << 16);
}

#define WMMA_BF16(A, B, C) \
    __builtin_amdgcn_wmma_f32_16x16x32_bf16(false, (A), false, (B), (short)0, (C), false, false)

// --------------------------------------------------------------------------
// 1) Per-row inverse norms: rn[row] = 1/sqrt(sum_c x^2), rows = 128*256, C=768
// --------------------------------------------------------------------------
__global__ __launch_bounds__(256)
void rownorm_kernel(const float* __restrict__ X, float* __restrict__ rn)
{
    const int row = blockIdx.x;
    const float* p = X + (size_t)row * 768;
    float s = 0.f;
    for (int i = threadIdx.x; i < 768; i += 256) { float v = p[i]; s += v * v; }
    __shared__ float red[256];
    red[threadIdx.x] = s; __syncthreads();
    for (int o = 128; o > 0; o >>= 1) {
        if (threadIdx.x < o) red[threadIdx.x] += red[threadIdx.x + o];
        __syncthreads();
    }
    if (threadIdx.x == 0) rn[row] = rsqrtf(red[0]);
}

// --------------------------------------------------------------------------
// 2) Normalized Gram (cosine self-similarity) via WMMA bf16 split.
//    grid = (4, 4, 128): 64x64 output tile per block, 256 threads = 8 waves.
//    Wave w: tile-row tm = w%4, tile-cols tn0 = 2*(w/4) .. tn0+1.
// --------------------------------------------------------------------------
__global__ __launch_bounds__(256)
void gram_ssm_kernel(const float* __restrict__ X, const float* __restrict__ rn,
                     float* __restrict__ ssm)
{
    const int b    = blockIdx.z;
    const int m0   = blockIdx.y * 64;
    const int n0   = blockIdx.x * 64;
    const int tid  = threadIdx.x;
    const int lane = tid & 31;
    const int wave = tid >> 5;
    const int half = (lane >> 4) & 1;
    const int lr   = lane & 15;
    const int tm   = wave & 3;
    const int tn0  = (wave >> 2) * 2;

    // bf16 pairs: uint index = row*16 + k/2 ; all gather ranges 16B aligned
    __shared__ __align__(16) unsigned Ahi[1024], Alo[1024], Bhi[1024], Blo[1024];

    const float* Xb = X + (size_t)b * (256 * 768);

    v8f acc0 = {}; v8f acc1 = {};

    for (int kb = 0; kb < 768; kb += 32) {
        __syncthreads();   // previous chunk's gathers done before restaging
        // ---- stage 64 rows x 32 K of A (rows m0..) and B (rows n0..) ----
        #pragma unroll
        for (int it = 0; it < 2; ++it) {
            const int q   = tid + it * 256;   // 0..511 float4 quads
            const int row = q >> 3;           // 0..63
            const int kq  = q & 7;            // 0..7
            const float* ap = Xb + (size_t)(m0 + row) * 768 + kb + kq * 4;
            const float* bp = Xb + (size_t)(n0 + row) * 768 + kb + kq * 4;
            if (kb + 32 < 768) {              // prefetch next chunk, keep near
                __builtin_prefetch(ap + 32, 0, 3);
                __builtin_prefetch(bp + 32, 0, 3);
            }
            const float4 av = *(const float4*)ap;
            const float4 bv = *(const float4*)bp;

            __bf16 ah0 = (__bf16)av.x, ah1 = (__bf16)av.y,
                   ah2 = (__bf16)av.z, ah3 = (__bf16)av.w;
            __bf16 al0 = (__bf16)(av.x - (float)ah0), al1 = (__bf16)(av.y - (float)ah1),
                   al2 = (__bf16)(av.z - (float)ah2), al3 = (__bf16)(av.w - (float)ah3);
            __bf16 bh0 = (__bf16)bv.x, bh1 = (__bf16)bv.y,
                   bh2 = (__bf16)bv.z, bh3 = (__bf16)bv.w;
            __bf16 bl0 = (__bf16)(bv.x - (float)bh0), bl1 = (__bf16)(bv.y - (float)bh1),
                   bl2 = (__bf16)(bv.z - (float)bh2), bl3 = (__bf16)(bv.w - (float)bh3);

            const int u = row * 16 + kq * 2;                // 8B aligned
            *(uint2*)&Ahi[u] = make_uint2(pack2(ah0, ah1), pack2(ah2, ah3));
            *(uint2*)&Alo[u] = make_uint2(pack2(al0, al1), pack2(al2, al3));
            *(uint2*)&Bhi[u] = make_uint2(pack2(bh0, bh1), pack2(bh2, bh3));
            *(uint2*)&Blo[u] = make_uint2(pack2(bl0, bl1), pack2(bl2, bl3));
        }
        __syncthreads();

        // ---- gather into WMMA register layouts (all ds_load_b128) ----
        // A 16x32: lanes0-15 hold K 0..15 ; lanes16-31 K 16..31 (row = lane%16)
        // B 32x16: lanes0-15 hold K 0..15 ; lanes16-31 K 16..31 (col = lane%16)
        BfVec a_hi, a_lo, b0_hi, b0_lo, b1_hi, b1_lo;
        const int arow  = (tm * 16 + lr) * 16;
        const int brow0 = (tn0 * 16 + lr) * 16;
        const int brow1 = ((tn0 + 1) * 16 + lr) * 16;

        a_hi.q[0]  = *(const uint4*)&Ahi[arow + 8 * half];
        a_hi.q[1]  = *(const uint4*)&Ahi[arow + 8 * half + 4];
        a_lo.q[0]  = *(const uint4*)&Alo[arow + 8 * half];
        a_lo.q[1]  = *(const uint4*)&Alo[arow + 8 * half + 4];
        b0_hi.q[0] = *(const uint4*)&Bhi[brow0 + 8 * half];
        b0_hi.q[1] = *(const uint4*)&Bhi[brow0 + 8 * half + 4];
        b0_lo.q[0] = *(const uint4*)&Blo[brow0 + 8 * half];
        b0_lo.q[1] = *(const uint4*)&Blo[brow0 + 8 * half + 4];
        b1_hi.q[0] = *(const uint4*)&Bhi[brow1 + 8 * half];
        b1_hi.q[1] = *(const uint4*)&Bhi[brow1 + 8 * half + 4];
        b1_lo.q[0] = *(const uint4*)&Blo[brow1 + 8 * half];
        b1_lo.q[1] = *(const uint4*)&Blo[brow1 + 8 * half + 4];

        // fp32-split product: hi*hi + hi*lo + lo*hi
        acc0 = WMMA_BF16(a_hi.v, b0_hi.v, acc0);
        acc0 = WMMA_BF16(a_hi.v, b0_lo.v, acc0);
        acc0 = WMMA_BF16(a_lo.v, b0_hi.v, acc0);
        acc1 = WMMA_BF16(a_hi.v, b1_hi.v, acc1);
        acc1 = WMMA_BF16(a_hi.v, b1_lo.v, acc1);
        acc1 = WMMA_BF16(a_lo.v, b1_hi.v, acc1);
    }

    // ---- normalize by 1/(|x_m||x_n|) and store ----
    const float* rnb = rn + b * 256;
    const float rn_n0 = rnb[n0 + tn0 * 16 + lr];
    const float rn_n1 = rnb[n0 + (tn0 + 1) * 16 + lr];
    float* out = ssm + (size_t)b * 65536;
    #pragma unroll
    for (int v = 0; v < 8; ++v) {
        const int m = m0 + tm * 16 + half * 8 + v;   // C layout: row = v + 8*(lane/16)
        const float rm = rnb[m];
        out[(size_t)m * 256 + n0 + tn0 * 16 + lr]       = acc0[v] * rm * rn_n0;
        out[(size_t)m * 256 + n0 + (tn0 + 1) * 16 + lr] = acc1[v] * rm * rn_n1;
    }
}

// --------------------------------------------------------------------------
// 3) waves[b,s] = mean_t ssm[b,t,s]      grid=128, block=256
// --------------------------------------------------------------------------
__global__ __launch_bounds__(256)
void waves_kernel(const float* __restrict__ ssm, float* __restrict__ w)
{
    const int b = blockIdx.x, s = threadIdx.x;
    const float* p = ssm + (size_t)b * 65536;
    float acc = 0.f;
    for (int t = 0; t < 256; ++t) acc += p[(size_t)t * 256 + s];
    w[b * 256 + s] = acc * (1.0f / 256.0f);
}

// --------------------------------------------------------------------------
// 4) Row-wise pearson (and optional sum of squared diffs for the MSE terms).
//    grid.x = nrows, block = 256.  sum(1-corr) -> pearson_acc.
// --------------------------------------------------------------------------
__global__ __launch_bounds__(256)
void pearson_rows_kernel(const float* __restrict__ X, const float* __restrict__ Y,
                         int rowlen, float* pearson_acc, float* sqdiff_acc)
{
    const int row = blockIdx.x;
    const int tid = threadIdx.x;
    const float* x = X + (size_t)row * rowlen;
    const float* y = Y + (size_t)row * rowlen;
    float sx = 0, sy = 0, sxx = 0, syy = 0, sxy = 0;
    for (int i = tid; i < rowlen; i += 256) {
        const float a = x[i], b = y[i];
        sx += a; sy += b; sxx += a * a; syy += b * b; sxy += a * b;
    }
    __shared__ float r0[256], r1[256], r2[256], r3[256], r4[256];
    r0[tid] = sx; r1[tid] = sy; r2[tid] = sxx; r3[tid] = syy; r4[tid] = sxy;
    __syncthreads();
    for (int o = 128; o > 0; o >>= 1) {
        if (tid < o) {
            r0[tid] += r0[tid + o]; r1[tid] += r1[tid + o]; r2[tid] += r2[tid + o];
            r3[tid] += r3[tid + o]; r4[tid] += r4[tid + o];
        }
        __syncthreads();
    }
    if (tid == 0) {
        const float n = (float)rowlen, inv = 1.f / (n - 1.f);
        const float cov = (r4[0] - r0[0] * r1[0] / n) * inv;
        const float vx  = fmaxf((r2[0] - r0[0] * r0[0] / n) * inv, 0.f);
        const float vy  = fmaxf((r3[0] - r1[0] * r1[0] / n) * inv, 0.f);
        float corr = cov / (sqrtf(vx) * sqrtf(vy) + 1e-7f);
        corr = fminf(1.f, fmaxf(-1.f, corr));
        atomicAdd(pearson_acc, 1.f - corr);
        if (sqdiff_acc)  // sum (x-y)^2 = sxx + syy - 2 sxy
            atomicAdd(sqdiff_acc, r2[0] + r3[0] - 2.f * r4[0]);
    }
}

// --------------------------------------------------------------------------
// 5) loss_reg superdiagonal stats: grid = (254, 128); k-th superdiagonal d=k+1
// --------------------------------------------------------------------------
__global__ __launch_bounds__(256)
void reg_kernel(const float* __restrict__ ssm, float* reg_acc)
{
    const int k = blockIdx.x, b = blockIdx.y, tid = threadIdx.x;
    const int d = k + 1, m = 255 - k;
    const float* p = ssm + (size_t)b * 65536;
    float s1 = 0, s2 = 0;
    for (int i = tid; i < m; i += 256) {
        const float v = p[(size_t)i * 256 + i + d];
        s1 += v; s2 += v * v;
    }
    __shared__ float r1[256], r2[256];
    r1[tid] = s1; r2[tid] = s2; __syncthreads();
    for (int o = 128; o > 0; o >>= 1) {
        if (tid < o) { r1[tid] += r1[tid + o]; r2[tid] += r2[tid + o]; }
        __syncthreads();
    }
    if (tid == 0) {
        const float fm = (float)m;                       // m >= 2 always here
        const float var = (r2[0] - r1[0] * r1[0] / fm) / (fm - 1.f);
        atomicAdd(reg_acc, sqrtf(fmaxf(var, 0.f)) * fm * 0.05f);
    }
}

// --------------------------------------------------------------------------
// 6) loss_freq: DFT bins 0..63 of waves (len 256); partition ratio is
//    scale-invariant so the max-normalization is dropped.  grid=128, block=64
// --------------------------------------------------------------------------
__global__ __launch_bounds__(64)
void freq_kernel(const float* __restrict__ w, float* freq_acc)
{
    const int b = blockIdx.x, f = threadIdx.x;
    const float* p = w + b * 256;
    const float step = -6.2831853071795864769f / 256.f;
    float re = 0, im = 0;
    for (int t = 0; t < 256; ++t) {
        const int ph = (f * t) & 255;
        float s, c; __sincosf(step * (float)ph, &s, &c);
        re += p[t] * c; im += p[t] * s;
    }
    __shared__ float ps[64];
    ps[f] = re * re + im * im;
    __syncthreads();
    if (f == 0) {
        float tot = 0, band = 0;
        for (int i = 0; i < 64; ++i) { tot += ps[i]; if (i >= 2 && i < 12) band += ps[i]; }
        atomicAdd(freq_acc, tot / band - 1.f);
    }
}

// --------------------------------------------------------------------------
// 7) rPPG PSD: DFT bins 0..287 of len-576 signal; store row (for the pearson
//    pass — pearson is scale-invariant so max-normalization is dropped) and
//    optionally accumulate the partition ratio.  grid=128, block=288
// --------------------------------------------------------------------------
__global__ __launch_bounds__(288)
void psd_rppg_kernel(const float* __restrict__ sig, float* __restrict__ out,
                     float* part_acc)
{
    const int b = blockIdx.x, f = threadIdx.x;
    const float* p = sig + b * 576;
    const float step = -6.2831853071795864769f / 576.f;
    float re = 0, im = 0;
    for (int t = 0; t < 576; ++t) {
        const int ph = (f * t) % 576;
        float s, c; __sincosf(step * (float)ph, &s, &c);
        re += p[t] * c; im += p[t] * s;
    }
    __shared__ float ps[288];
    const float val = re * re + im * im;
    ps[f] = val;
    out[b * 288 + f] = val;
    __syncthreads();
    if (f == 0 && part_acc) {
        float tot = 0, band = 0;
        for (int i = 0; i < 288; ++i) { tot += ps[i]; if (i >= 9 && i < 57) band += ps[i]; }
        atomicAdd(part_acc, tot / band - 1.f);
    }
}

// --------------------------------------------------------------------------
// 8) Combine accumulators into the 5 output scalars
// --------------------------------------------------------------------------
__global__ void finalize_kernel(const float* __restrict__ acc, float* __restrict__ out)
{
    if (threadIdx.x == 0) {
        const float mse = acc[1] / (128.f * 65536.f) + acc[3] / (128.f * 256.f);
        const float loss_pyr  = 0.5f * (mse + acc[0] / 32768.f + acc[2] / 128.f);
        const float loss_r    = 0.5f * (acc[4] / (128.f * 254.f));
        const float loss_f    = 0.2f * (acc[5] / 128.f);
        const float loss_rppg = 0.5f * ((acc[6] + acc[7] + acc[8]) / 128.f);
        out[0] = loss_rppg + loss_pyr + loss_r + loss_f;
        out[1] = loss_pyr;
        out[2] = loss_rppg;
        out[3] = loss_r;
        out[4] = loss_f;
    }
}

// --------------------------------------------------------------------------
extern "C" void kernel_launch(void* const* d_in, const int* in_sizes, int n_in,
                              void* d_out, int out_size, void* d_ws, size_t ws_size,
                              hipStream_t stream)
{
    (void)in_sizes; (void)n_in; (void)out_size; (void)ws_size;
    const float* pyr_on = (const float*)d_in[0];   // 128*256*768
    const float* pyr_tg = (const float*)d_in[1];
    const float* rp_on  = (const float*)d_in[2];   // 128*576
    const float* rp_tg  = (const float*)d_in[3];

    float* ws     = (float*)d_ws;
    float* acc    = ws;                        // 16 accumulator slots
    float* rn_on  = ws + 16;                   // 32768
    float* rn_tg  = rn_on + 32768;             // 32768
    float* on_ssm = rn_tg + 32768;             // 8388608
    float* tg_ssm = on_ssm + 8388608;          // 8388608
    float* w_on   = tg_ssm + 8388608;          // 32768
    float* w_tg   = w_on + 32768;              // 32768
    float* so     = w_tg + 32768;              // 36864
    float* st     = so + 36864;                // 36864   (~68 MB total)

    hipMemsetAsync(acc, 0, 16 * sizeof(float), stream);

    rownorm_kernel<<<32768, 256, 0, stream>>>(pyr_on, rn_on);
    rownorm_kernel<<<32768, 256, 0, stream>>>(pyr_tg, rn_tg);

    dim3 gg(4, 4, 128);
    gram_ssm_kernel<<<gg, 256, 0, stream>>>(pyr_on, rn_on, on_ssm);
    gram_ssm_kernel<<<gg, 256, 0, stream>>>(pyr_tg, rn_tg, tg_ssm);

    waves_kernel<<<128, 256, 0, stream>>>(on_ssm, w_on);
    waves_kernel<<<128, 256, 0, stream>>>(tg_ssm, w_tg);

    pearson_rows_kernel<<<32768, 256, 0, stream>>>(on_ssm, tg_ssm, 256, acc + 0, acc + 1);
    pearson_rows_kernel<<<128,   256, 0, stream>>>(w_on,   w_tg,   256, acc + 2, acc + 3);

    reg_kernel<<<dim3(254, 128), 256, 0, stream>>>(on_ssm, acc + 4);
    freq_kernel<<<128, 64, 0, stream>>>(w_on, acc + 5);

    pearson_rows_kernel<<<128, 256, 0, stream>>>(rp_on, rp_tg, 576, acc + 6, nullptr);
    psd_rppg_kernel<<<128, 288, 0, stream>>>(rp_on, so, acc + 8);
    psd_rppg_kernel<<<128, 288, 0, stream>>>(rp_tg, st, nullptr);
    pearson_rows_kernel<<<128, 256, 0, stream>>>(so, st, 288, acc + 7, nullptr);

    finalize_kernel<<<1, 32, 0, stream>>>(acc, (float*)d_out);
}
